// Knrm_60619168415748
// MI455X (gfx1250) — matrix operand order repaired and checked by reading, not compiled, not run on hardware
//
#include <hip/hip_runtime.h>

// KNRM: B=256, Q=16, D=512, E=300, V=100000, NB=11 bins.
// sim GEMM done in f16 WMMA (v_wmma_f32_16x16x32_f16), RBF pooling in f32.

#define B_SZ 256
#define Q_SZ 16
#define D_SZ 512
#define E_SZ 300
#define KP   320      // K padded to 10 x 32
#define NB   11
#define DT   64       // doc-tile per block
#define NW   4        // waves per main block (one 16-doc subtile each)

typedef __attribute__((ext_vector_type(16))) _Float16 v16h;
typedef __attribute__((ext_vector_type(8)))  _Float16 v8h;
typedef __attribute__((ext_vector_type(8)))  float    v8f;

union AFrag { v16h v; v8h h[2]; };

__global__ void knrm_zero(float* __restrict__ p, int n) {
    int i = blockIdx.x * blockDim.x + threadIdx.x;
    if (i < n) p[i] = 0.0f;
}

// One wave (32 lanes) per query row: gather embedding row, L2-normalize,
// store f16 padded to KP.
__global__ void knrm_prep_q(const int* __restrict__ inputs_q,
                            const float* __restrict__ emb,
                            _Float16* __restrict__ q16) {
    const int wave = threadIdx.x >> 5;
    const int lane = threadIdx.x & 31;
    const int row  = blockIdx.x * 4 + wave;     // 0 .. B*Q-1
    if (row >= B_SZ * Q_SZ) return;

    const float* src = emb + (long)inputs_q[row] * E_SZ;
    float v[10];
    float ss = 0.0f;
#pragma unroll
    for (int i = 0; i < 10; ++i) {
        int k = lane + 32 * i;
        float x = (k < E_SZ) ? src[k] : 0.0f;
        v[i] = x;
        ss += x * x;
    }
#pragma unroll
    for (int off = 16; off > 0; off >>= 1) ss += __shfl_xor(ss, off, 32);
    const float inv = 1.0f / (sqrtf(ss) + 1e-10f);

    _Float16* dst = q16 + (long)row * KP;
#pragma unroll
    for (int i = 0; i < 10; ++i) {
        int k = lane + 32 * i;
        dst[k] = (_Float16)((k < E_SZ) ? v[i] * inv : 0.0f);
    }
}

// grid = (D_SZ/DT, B_SZ); block = NW*32 threads.
// LDS: doc tile DT x KP f16 (normalized) + query tile Q_SZ x KP f16.
__global__ __launch_bounds__(NW * 32, 1)
void knrm_main(const int* __restrict__ inputs_d,
               const float* __restrict__ emb,
               const float* __restrict__ mask,
               const float* __restrict__ mu,
               const float* __restrict__ sigma,
               const _Float16* __restrict__ q16,
               float* __restrict__ kdeAcc) {
    extern __shared__ __attribute__((aligned(32))) _Float16 smem[];
    _Float16* sD = smem;            // DT * KP
    _Float16* sQ = smem + DT * KP;  // Q_SZ * KP

    const int dt   = blockIdx.x;    // doc tile
    const int b    = blockIdx.y;    // batch
    const int tid  = threadIdx.x;
    const int wave = tid >> 5;
    const int lane = tid & 31;

    // RBF parameters in registers.
    float mu_r[NB], nis_r[NB];
#pragma unroll
    for (int k = 0; k < NB; ++k) {
        mu_r[k] = mu[k];
        float s = sigma[k];
        nis_r[k] = -1.0f / (2.0f * s * s);
    }

    // Load normalized query tile (contiguous f16 copy).
    {
        const _Float16* src = q16 + (long)b * Q_SZ * KP;
        for (int i = tid; i < Q_SZ * KP; i += NW * 32) sQ[i] = src[i];
    }

    // Gather + normalize DT doc rows into LDS (each wave owns DT/NW rows).
    for (int it = 0; it < DT / NW; ++it) {
        const int r    = wave * (DT / NW) + it;
        const int dRow = b * D_SZ + dt * DT + r;
        const float* src = emb + (long)inputs_d[dRow] * E_SZ;
        float v[10];
        float ss = 0.0f;
#pragma unroll
        for (int i = 0; i < 10; ++i) {
            int k = lane + 32 * i;
            float x = (k < E_SZ) ? src[k] : 0.0f;
            v[i] = x;
            ss += x * x;
        }
#pragma unroll
        for (int off = 16; off > 0; off >>= 1) ss += __shfl_xor(ss, off, 32);
        const float inv = 1.0f / (sqrtf(ss) + 1e-10f);
        _Float16* dst = sD + (long)r * KP;
#pragma unroll
        for (int i = 0; i < 10; ++i) {
            int k = lane + 32 * i;
            dst[k] = (_Float16)((k < E_SZ) ? v[i] * inv : 0.0f);
        }
    }
    __syncthreads();

    // WMMA: each wave computes a 16(Q) x 16(doc) sim tile over K=KP.
    // 16-bit A 16x32 order used here (contiguous k, not the ISA table order): lane half = lane>=16; per lane two
    // contiguous 8xf16 chunks at K = kb + half*16 (+8 for VGPR4..7).
    // B 32x16 layout: lane = column N (mod 16), K = kb + half*16 + {0..15}.
    const int half  = lane >> 4;
    const int l15   = lane & 15;
    const int nBase = wave * 16;

    v8f c = {};
#pragma unroll
    for (int kb = 0; kb < KP / 32; ++kb) {
        const int kBase = kb * 32;
        AFrag a;
        const _Float16* ap = sQ + l15 * KP + kBase + half * 16;
        a.h[0] = *(const v8h*)(ap);
        a.h[1] = *(const v8h*)(ap + 8);
        const _Float16* bp = sD + (nBase + l15) * KP + kBase + half * 16;
        v16h bv = *(const v16h*)bp;
        c = __builtin_amdgcn_wmma_f32_16x16x32_f16(
                /*neg_a=*/false, a.v, /*neg_b=*/false, bv,
                /*c_mod=*/(short)0, c, /*reuse_a=*/false, /*reuse_b=*/false);
    }

    // C/D layout: VGPR r holds M = r + 8*half, N = lane&15.
    // RBF per bin, reduce over the 16 lanes sharing an M row (sum over docs),
    // accumulate into kdeAcc[b][m][k] with float atomics.
    const int dGlob = dt * DT + nBase + l15;
#pragma unroll
    for (int r = 0; r < 8; ++r) {
        const int m   = r + 8 * half;
        const float sim = c[r];
        const float mk  = mask[((long)b * Q_SZ + m) * D_SZ + dGlob];
#pragma unroll
        for (int k = 0; k < NB; ++k) {
            float d   = sim - mu_r[k];
            float val = mk * __expf(d * d * nis_r[k]);
#pragma unroll
            for (int off = 8; off > 0; off >>= 1) val += __shfl_xor(val, off, 32);
            if (l15 == 0)
                atomicAdd(&kdeAcc[((long)b * Q_SZ + m) * NB + k], val);
        }
    }
}

// One 32-lane block per batch: log soft-TF, q-weighted sum, dense+tanh.
__global__ void knrm_finalize(const float* __restrict__ kdeAcc,
                              const float* __restrict__ q_weights,
                              const float* __restrict__ W1,
                              const float* __restrict__ b1,
                              float* __restrict__ out) {
    const int b    = blockIdx.x;
    const int lane = threadIdx.x;
    float agg = 0.0f;
    if (lane < NB) {
        for (int q = 0; q < Q_SZ; ++q) {
            float kv = kdeAcc[((long)b * Q_SZ + q) * NB + lane];
            kv = __logf(fmaxf(kv, 1e-10f)) * 0.01f;
            agg += kv * q_weights[b * Q_SZ + q];
        }
        agg *= W1[lane];
    }
#pragma unroll
    for (int off = 16; off > 0; off >>= 1) agg += __shfl_xor(agg, off, 32);
    if (lane == 0) out[b] = tanhf(agg + b1[0]);
}

extern "C" void kernel_launch(void* const* d_in, const int* in_sizes, int n_in,
                              void* d_out, int out_size, void* d_ws, size_t ws_size,
                              hipStream_t stream) {
    const int*   inputs_q  = (const int*)d_in[0];
    const int*   inputs_d  = (const int*)d_in[1];
    const float* mask      = (const float*)d_in[2];
    const float* q_weights = (const float*)d_in[3];
    const float* mu        = (const float*)d_in[4];
    const float* sigma     = (const float*)d_in[5];
    const float* emb       = (const float*)d_in[6];
    const float* W1        = (const float*)d_in[7];
    const float* b1        = (const float*)d_in[8];
    float* out = (float*)d_out;

    // Workspace layout: [q16 f16 B*Q*KP][kdeAcc f32 B*Q*NB]  (~2.8 MB)
    _Float16* q16    = (_Float16*)d_ws;
    float*    kdeAcc = (float*)((char*)d_ws +
                                (size_t)B_SZ * Q_SZ * KP * sizeof(_Float16));

    const int accN = B_SZ * Q_SZ * NB;
    hipLaunchKernelGGL(knrm_zero, dim3((accN + 255) / 256), dim3(256), 0, stream,
                       kdeAcc, accN);

    hipLaunchKernelGGL(knrm_prep_q, dim3(B_SZ * Q_SZ / 4), dim3(128), 0, stream,
                       inputs_q, emb, q16);

    const size_t smemBytes = (size_t)(DT + Q_SZ) * KP * sizeof(_Float16); // 51200
    hipLaunchKernelGGL(knrm_main, dim3(D_SZ / DT, B_SZ), dim3(NW * 32),
                       smemBytes, stream,
                       inputs_d, emb, mask, mu, sigma, q16, kdeAcc);

    hipLaunchKernelGGL(knrm_finalize, dim3(B_SZ), dim3(32), 0, stream,
                       kdeAcc, q_weights, W1, b1, out);
}
